// SelfAttention3D_51513837748901
// MI455X (gfx1250) — matrix operand order repaired and checked by reading, not compiled, not run on hardware
//
#include <hip/hip_runtime.h>
#include <hip/hip_bf16.h>

typedef _Float16 half_t;
typedef __attribute__((ext_vector_type(16))) _Float16 v16h;
typedef __attribute__((ext_vector_type(8)))  _Float16 v8h;
typedef __attribute__((ext_vector_type(8)))  float    v8f;

#define B_   2
#define C_   128
#define S_   4096
#define NH_  4
#define HD_  32
#define GRP_ 8
#define SCALE_Q 0.17677669529663687f   // 32^-0.5

#define WMMA_F16(a, b, c) \
    __builtin_amdgcn_wmma_f32_16x16x32_f16(false, (a), false, (b), (short)0, (c), false, false)

// ---------------- WMMA fragment loaders (ISA 7.12.2 layouts, wave32) ----------

// A fragment: 16(M) x 32(K) f16 tile, row-major, stride lda (halves).
__device__ __forceinline__ v16h load_a_frag(const half_t* A, int lda) {
    const int lane = threadIdx.x & 31;
    const int m = lane & 15;
    const int koff = (lane >> 4) * 8;
    v8h lo = *(const v8h*)(A + (size_t)m * lda + koff);
    v8h hi = *(const v8h*)(A + (size_t)m * lda + 16 + koff);
    v16h a;
#pragma unroll
    for (int j = 0; j < 8; ++j) { a[j] = lo[j]; a[j + 8] = hi[j]; }
    return a;
}

// B fragment from transposed (N-major, K-contiguous) storage.
// lane: n = lane&15 ; halves j<8 -> K = (lane>>4)*8 + j, j>=8 -> K = 16 + (lane>>4)*8 + (j-8).
__device__ __forceinline__ v16h load_bT_frag(const half_t* BT, int ldt) {
    const int lane = threadIdx.x & 31;
    const int n = lane & 15;
    const int kbase = (lane >> 4) * 8;
    v8h lo = *(const v8h*)(BT + (size_t)n * ldt + kbase);
    v8h hi = *(const v8h*)(BT + (size_t)n * ldt + kbase + 16);
    v16h b;
#pragma unroll
    for (int j = 0; j < 8; ++j) { b[j] = lo[j]; b[j + 8] = hi[j]; }
    return b;
}

// C/D layout: element (m, n): n = lane&15, m = r + 8*(lane>>4), r = vgpr 0..7

// ---------------- 1) GroupNorm statistics -------------------------------------
__global__ void gn_stats_kernel(const float* __restrict__ x,
                                float* __restrict__ mu, float* __restrict__ rstd) {
    const int bg = blockIdx.x;                 // 0..15 (B*GROUPS)
    const float4* p = (const float4*)(x + (size_t)bg * 65536);
    float s = 0.f, s2 = 0.f;
    for (int i = threadIdx.x; i < 16384; i += 256) {
        float4 v = p[i];
        s  += v.x + v.y + v.z + v.w;
        s2 += v.x * v.x + v.y * v.y + v.z * v.z + v.w * v.w;
    }
#pragma unroll
    for (int msk = 16; msk; msk >>= 1) {
        s  += __shfl_xor(s, msk, 32);
        s2 += __shfl_xor(s2, msk, 32);
    }
    __shared__ float sh[16];
    const int wave = threadIdx.x >> 5, lane = threadIdx.x & 31;
    if (lane == 0) { sh[wave] = s; sh[8 + wave] = s2; }
    __syncthreads();
    if (threadIdx.x == 0) {
        float ts = 0.f, ts2 = 0.f;
        for (int w = 0; w < 8; ++w) { ts += sh[w]; ts2 += sh[8 + w]; }
        float m = ts * (1.f / 65536.f);
        float var = ts2 * (1.f / 65536.f) - m * m;
        mu[bg] = m;
        rstd[bg] = rsqrtf(var + 1e-5f);
    }
}

// ------- 2) apply GN, pack h transposed to f16: hT[b][s][c] -------------------
__global__ void pack_h_kernel(const float* __restrict__ x,
                              const float* __restrict__ gn_w, const float* __restrict__ gn_b,
                              const float* __restrict__ mu, const float* __restrict__ rstd,
                              half_t* __restrict__ hT) {
    const int idx = blockIdx.x * 256 + threadIdx.x;        // 0..262143 : (b, s, c/4)
    const int c = (idx & 31) * 4;
    const int s = (idx >> 5) & (S_ - 1);
    const int b = idx >> 17;
    half_t* dst = hT + ((size_t)b * S_ + s) * C_ + c;
#pragma unroll
    for (int j = 0; j < 4; ++j) {
        const int cc = c + j;
        const int bg = b * GRP_ + (cc >> 4);               // 16 channels per group
        const float xv = x[((size_t)b * C_ + cc) * S_ + s];
        dst[j] = (half_t)((xv - mu[bg]) * rstd[bg] * gn_w[cc] + gn_b[cc]);
    }
}

// ---------------- 3) weights to f16 -------------------------------------------
__global__ void pack_w_kernel(const float* __restrict__ qkv_w, const float* __restrict__ proj_w,
                              half_t* __restrict__ wq, half_t* __restrict__ wp) {
    const int i = blockIdx.x * 256 + threadIdx.x;          // 0..65535
    if (i < 3 * C_ * C_) wq[i] = (half_t)qkv_w[i];
    else                 wp[i - 3 * C_ * C_] = (half_t)proj_w[i - 3 * C_ * C_];
}

// ---------------- 4) QKV GEMM -------------------------------------------------
// one wave per 16x16 tile; q:[bh,s,d]*scale, k:[bh,t,d] (natural), vT:[bh,d,t]
// K-loop uses two independent accumulator chains to avoid WMMA RAW hazard NOPs.
__global__ void qkv_gemm_kernel(const half_t* __restrict__ wq, const half_t* __restrict__ hT,
                                const float* __restrict__ qkv_b,
                                half_t* __restrict__ q, half_t* __restrict__ k,
                                half_t* __restrict__ vT) {
    const int wave = threadIdx.x >> 5, lane = threadIdx.x & 31;
    const int gw = blockIdx.x * 8 + wave;          // 0..12287
    const int b = (gw >= 6144) ? 1 : 0;
    const int t = gw - b * 6144;                   // 24 o-tiles * 256 s-tiles
    const int o0 = (t >> 8) * 16;
    const int s0 = (t & 255) * 16;
    const half_t* hb = hT + ((size_t)b * S_ + s0) * C_;
    v8f acc0 = {}, acc1 = {};
#pragma unroll
    for (int k0 = 0; k0 < C_; k0 += 64) {
        v16h a0 = load_a_frag(wq + (size_t)o0 * C_ + k0, C_);
        v16h b0 = load_bT_frag(hb + k0, C_);
        v16h a1 = load_a_frag(wq + (size_t)o0 * C_ + k0 + 32, C_);
        v16h b1 = load_bT_frag(hb + k0 + 32, C_);
        acc0 = WMMA_F16(a0, b0, acc0);
        acc1 = WMMA_F16(a1, b1, acc1);
    }
    v8f acc;
#pragma unroll
    for (int r = 0; r < 8; ++r) acc[r] = acc0[r] + acc1[r];

    const int n = lane & 15, hi = lane >> 4;
    const int s = s0 + n;
    // tile-level routing is wave-uniform: a 16-row o-tile never crosses q/k/v
    const int which = o0 >> 7;                     // 0=q 1=k 2=v (uniform)
    const int oc0 = o0 & 127;
    const size_t bh = (size_t)(b * NH_ + (oc0 >> 5));
    const int d0 = (oc0 & 31) + 8 * hi;            // this lane's 8 d's: d0..d0+7
    if (which == 0) {
        v8h pk;
#pragma unroll
        for (int r = 0; r < 8; ++r)
            pk[r] = (half_t)((acc[r] + qkv_b[o0 + 8 * hi + r]) * SCALE_Q);
        *(v8h*)(q + (bh * S_ + s) * HD_ + d0) = pk;
    } else if (which == 1) {
        v8h pk;
#pragma unroll
        for (int r = 0; r < 8; ++r)
            pk[r] = (half_t)(acc[r] + qkv_b[o0 + 8 * hi + r]);
        *(v8h*)(k + (bh * S_ + s) * HD_ + d0) = pk;
    } else {
#pragma unroll
        for (int r = 0; r < 8; ++r)
            vT[(bh * HD_ + d0 + r) * S_ + s] =
                (half_t)(acc[r] + qkv_b[o0 + 8 * hi + r]);
    }
}

// ---------------- 5) flash attention ------------------------------------------
// 8 waves/block share one (b,head); k/v tiles staged to LDS via async copy,
// double-buffered. q,k: [bh][s][32]; vT: [bh][32][t]; out aoT: [b][s][c].
__global__ void attn_kernel(const half_t* __restrict__ qb, const half_t* __restrict__ kb,
                            const half_t* __restrict__ vTb, half_t* __restrict__ aoT) {
    __shared__ half_t kstage[2][64 * HD_];         // 64 t-rows x 32 d   (4 KB each)
    __shared__ half_t vstage[2][HD_ * 64];         // 32 d-rows x 64 t   (4 KB each)
    __shared__ half_t plds[8][16 * 32];            // per-wave P staging (8 KB)
    const int tid = threadIdx.x;
    const int wave = tid >> 5, lane = tid & 31;
    const int gw = blockIdx.x * 8 + wave;          // 0..2047
    const int s0 = (gw & 255) * 16;
    const int bh = gw >> 8;                        // uniform across block (256 s-tiles/bh)
    const half_t* qp  = qb  + (size_t)bh * S_ * HD_;
    const half_t* kp  = kb  + (size_t)bh * S_ * HD_;
    const half_t* vTp = vTb + (size_t)bh * HD_ * S_;

    const v16h qa = load_a_frag(qp + (size_t)s0 * HD_, HD_);   // 16 s-rows x 32 d

    // async stage of one 64-wide t-block: k tile is 4KB contiguous, v tile is
    // 32 rows of 128B. Each thread moves one 16B chunk of each (2 async/wave).
    auto issue_stage = [&](int buf, int t0) {
        const char* kg = (const char*)(kp + (size_t)t0 * HD_) + tid * 16;
        unsigned    kl = (unsigned)(size_t)(&kstage[buf][0]) + tid * 16;
        asm volatile("global_load_async_to_lds_b128 %0, %1, off"
                     :: "v"(kl), "v"(kg) : "memory");
        const int d = tid >> 3, off = (tid & 7) * 16;
        const char* vg = (const char*)vTp + (size_t)d * (S_ * 2) + (size_t)t0 * 2 + off;
        unsigned    vl = (unsigned)(size_t)(&vstage[buf][0]) + tid * 16;
        asm volatile("global_load_async_to_lds_b128 %0, %1, off"
                     :: "v"(vl), "v"(vg) : "memory");
    };

    float mrow[8], lrow[8];
#pragma unroll
    for (int r = 0; r < 8; ++r) { mrow[r] = -3.0e38f; lrow[r] = 0.f; }
    v8f o0 = {}, o1 = {};
    half_t* pl = plds[wave];
    const int n = lane & 15, hi = lane >> 4;

    issue_stage(0, 0);
    int cur = 0;
    for (int t0 = 0; t0 < S_; t0 += 64) {
        const bool more = (t0 + 64 < S_);
        if (more) {
            issue_stage(cur ^ 1, t0 + 64);
            asm volatile("s_wait_asynccnt 0x2" ::: "memory");   // cur's 2 done
        } else {
            asm volatile("s_wait_asynccnt 0x0" ::: "memory");
        }
        __syncthreads();                                        // all waves staged
        const half_t* kt = &kstage[cur][0];                     // [t][d], ld 32
        const half_t* vt = &vstage[cur][0];                     // [d][t], ld 64

        // scores: q(16x32d) x k^T tiles -> four 16-col tiles (64 t-columns)
        v8f c[4];
#pragma unroll
        for (int u = 0; u < 4; ++u) {
            v16h kf = load_bT_frag(kt + 16 * u * HD_, HD_);
            v8f z = {};
            c[u] = WMMA_F16(qa, kf, z);
        }
        // online softmax across 64 columns
        float alpha[8];
#pragma unroll
        for (int r = 0; r < 8; ++r) {
            float mx = fmaxf(fmaxf(c[0][r], c[1][r]), fmaxf(c[2][r], c[3][r]));
            mx = fmaxf(mx, __shfl_xor(mx, 1, 32));
            mx = fmaxf(mx, __shfl_xor(mx, 2, 32));
            mx = fmaxf(mx, __shfl_xor(mx, 4, 32));
            mx = fmaxf(mx, __shfl_xor(mx, 8, 32));
            const float nm = fmaxf(mrow[r], mx);
            alpha[r] = __expf(mrow[r] - nm);
            mrow[r] = nm;
            c[0][r] = __expf(c[0][r] - nm);
            c[1][r] = __expf(c[1][r] - nm);
            c[2][r] = __expf(c[2][r] - nm);
            c[3][r] = __expf(c[3][r] - nm);
            float rs = (c[0][r] + c[1][r]) + (c[2][r] + c[3][r]);
            rs += __shfl_xor(rs, 1, 32);
            rs += __shfl_xor(rs, 2, 32);
            rs += __shfl_xor(rs, 4, 32);
            rs += __shfl_xor(rs, 8, 32);
            lrow[r] = lrow[r] * alpha[r] + rs;
            o0[r] *= alpha[r];
            o1[r] *= alpha[r];
        }
        // two P half-blocks (16x32 each): C-layout -> LDS -> A fragment, then PV
#pragma unroll
        for (int hb2 = 0; hb2 < 2; ++hb2) {
#pragma unroll
            for (int r = 0; r < 8; ++r) {
                const int m = r + 8 * hi;
                pl[m * 32 + n]      = (half_t)c[2 * hb2][r];
                pl[m * 32 + 16 + n] = (half_t)c[2 * hb2 + 1][r];
            }
            asm volatile("s_wait_dscnt 0" ::: "memory");
            v16h pa = load_a_frag(pl, 32);
            // V tile rows t in [32*hb2, +32) of stage; cols d: two 16-wide tiles
            v16h vf0 = load_bT_frag(vt + 32 * hb2, 64);
            v16h vf1 = load_bT_frag(vt + 16 * 64 + 32 * hb2, 64);
            o0 = WMMA_F16(pa, vf0, o0);
            o1 = WMMA_F16(pa, vf1, o1);
            asm volatile("s_wait_dscnt 0" ::: "memory");
        }
        __syncthreads();                // all waves done reading before overwrite
        cur ^= 1;
    }

    // normalize + write attn-out transposed: aoT[b][s][c], c = head*32 + d
    const int b = bh >> 2, head = bh & 3;
#pragma unroll
    for (int r = 0; r < 8; ++r) {
        const int m = r + 8 * hi;                  // s row
        const float inv = 1.f / lrow[r];
        half_t* dst = aoT + ((size_t)b * S_ + s0 + m) * C_ + head * HD_;
        dst[n]      = (half_t)(o0[r] * inv);
        dst[16 + n] = (half_t)(o1[r] * inv);
    }
}

// ---------------- 6) proj GEMM + bias + residual ------------------------------
__global__ void proj_gemm_kernel(const half_t* __restrict__ wp, const half_t* __restrict__ aoT,
                                 const float* __restrict__ proj_b, const float* __restrict__ x,
                                 float* __restrict__ out) {
    const int wave = threadIdx.x >> 5, lane = threadIdx.x & 31;
    const int gw = blockIdx.x * 8 + wave;          // 0..4095
    const int b = (gw >= 2048) ? 1 : 0;
    const int t = gw - b * 2048;                   // 8 o-tiles * 256 s-tiles
    const int o0 = (t >> 8) * 16;
    const int s0 = (t & 255) * 16;
    const half_t* aob = aoT + ((size_t)b * S_ + s0) * C_;
    v8f acc0 = {}, acc1 = {};
#pragma unroll
    for (int k0 = 0; k0 < C_; k0 += 64) {
        v16h a0 = load_a_frag(wp + (size_t)o0 * C_ + k0, C_);
        v16h b0 = load_bT_frag(aob + k0, C_);
        v16h a1 = load_a_frag(wp + (size_t)o0 * C_ + k0 + 32, C_);
        v16h b1 = load_bT_frag(aob + k0 + 32, C_);
        acc0 = WMMA_F16(a0, b0, acc0);
        acc1 = WMMA_F16(a1, b1, acc1);
    }
    const int n = lane & 15, hi = lane >> 4;
    const int s = s0 + n;
#pragma unroll
    for (int r = 0; r < 8; ++r) {
        const int o = o0 + r + 8 * hi;
        const size_t idx = (size_t)(b * C_ + o) * S_ + s;
        out[idx] = x[idx] + proj_b[o] + (acc0[r] + acc1[r]);
    }
}

// ---------------- host launcher ----------------------------------------------
extern "C" void kernel_launch(void* const* d_in, const int* in_sizes, int n_in,
                              void* d_out, int out_size, void* d_ws, size_t ws_size,
                              hipStream_t stream) {
    const float* x      = (const float*)d_in[0];
    const float* gn_w   = (const float*)d_in[1];
    const float* gn_b   = (const float*)d_in[2];
    const float* qkv_w  = (const float*)d_in[3];
    const float* qkv_b  = (const float*)d_in[4];
    const float* proj_w = (const float*)d_in[5];
    const float* proj_b = (const float*)d_in[6];
    float* out = (float*)d_out;

    char* ws = (char*)d_ws;
    const size_t HB = (size_t)B_ * C_ * S_ * sizeof(half_t);   // 2 MiB
    float*  mu   = (float*)(ws + 0);
    float*  rstd = (float*)(ws + 128);
    half_t* hT   = (half_t*)(ws + 256);
    half_t* wq16 = (half_t*)(ws + 256 + HB);
    half_t* wp16 = (half_t*)(ws + 256 + HB + 3 * C_ * C_ * sizeof(half_t));
    char*   base = (char*)wp16 + (size_t)C_ * C_ * sizeof(half_t);
    half_t* qbuf = (half_t*)(base);
    half_t* kbuf = (half_t*)(base + HB);
    half_t* vT   = (half_t*)(base + 2 * HB);
    half_t* aoT  = (half_t*)(base + 3 * HB);

    gn_stats_kernel<<<B_ * GRP_, 256, 0, stream>>>(x, mu, rstd);
    pack_h_kernel<<<1024, 256, 0, stream>>>(x, gn_w, gn_b, mu, rstd, hT);
    pack_w_kernel<<<(4 * C_ * C_) / 256, 256, 0, stream>>>(qkv_w, proj_w, wq16, wp16);
    qkv_gemm_kernel<<<1536, 256, 0, stream>>>(wq16, hT, qkv_b, qbuf, kbuf, vT);
    attn_kernel<<<256, 256, 0, stream>>>(qbuf, kbuf, vT, aoT);
    proj_gemm_kernel<<<512, 256, 0, stream>>>(wp16, aoT, proj_b, x, out);
}
